// ChebModel_74380243632480
// MI455X (gfx1250) — matrix operand order repaired and checked by reading, not compiled, not run on hardware
//
#include <hip/hip_runtime.h>

#define NODES  50000
#define EDGES  600000
#define FEAT   128
#define HID    512
#define NOUT   128
#define NGRAPH 128

typedef __attribute__((ext_vector_type(16))) __bf16 bf16x16;
typedef __attribute__((ext_vector_type(8)))  float  v8f;

__device__ __forceinline__ unsigned short f2bf(float f) {
    union { float f; unsigned u; } v; v.f = f;
    unsigned u = v.u;
    return (unsigned short)((u + 0x7FFFu + ((u >> 16) & 1u)) >> 16); // RNE
}

// ---------------- setup: degrees, dinv, edge weights ----------------
__global__ void deg_kernel(const int* __restrict__ src, float* __restrict__ deg) {
    int e = blockIdx.x * blockDim.x + threadIdx.x;
    if (e < EDGES) atomicAdd(&deg[src[e]], 1.0f);
}

__global__ void dinv_kernel(float* __restrict__ deg) {
    int i = blockIdx.x * blockDim.x + threadIdx.x;
    if (i < NODES) {
        float d = deg[i];
        deg[i] = (d > 0.f) ? rsqrtf(fmaxf(d, 1.f)) : 0.f;
    }
}

__global__ void edgew_kernel(const int* __restrict__ src, const int* __restrict__ dst,
                             const float* __restrict__ dinv, float* __restrict__ ew) {
    int e = blockIdx.x * blockDim.x + threadIdx.x;
    if (e < EDGES) ew[e] = -dinv[src[e]] * dinv[dst[e]];
}

// ---------------- sparse propagation: out[dst] += w * h[src] ----------------
// thread = (edge, 4-float chunk); FEAT=128 -> 32 chunks/edge
__global__ void prop_kernel(const float* __restrict__ h, const float* __restrict__ ew,
                            const int* __restrict__ src, const int* __restrict__ dst,
                            float* __restrict__ out) {
    int idx = blockIdx.x * blockDim.x + threadIdx.x;
    if (idx >= EDGES * 32) return;
    int e = idx >> 5;
    int c = (idx & 31) << 2;
    float w = ew[e];
    int s = src[e], d = dst[e];
    float4 v = *(const float4*)(h + (size_t)s * FEAT + c);
    float* o = out + (size_t)d * FEAT + c;
    atomicAdd(o + 0, w * v.x);
    atomicAdd(o + 1, w * v.y);
    atomicAdd(o + 2, w * v.z);
    atomicAdd(o + 3, w * v.w);
}

// ---------------- conversions ----------------
__global__ void cvt_bf16_kernel(const float* __restrict__ x, unsigned short* __restrict__ y, int n4) {
    int i = blockIdx.x * blockDim.x + threadIdx.x;
    if (i < n4) {
        float4 v = ((const float4*)x)[i];
        ushort4 o;
        o.x = f2bf(v.x); o.y = f2bf(v.y); o.z = f2bf(v.z); o.w = f2bf(v.w);
        ((ushort4*)y)[i] = o;
    }
}

// Chebyshev T2 = 2*prop(T1) - T0, emitted directly as bf16
__global__ void cheb2_kernel(const float* __restrict__ p, const float* __restrict__ x0,
                             unsigned short* __restrict__ y, int n4) {
    int i = blockIdx.x * blockDim.x + threadIdx.x;
    if (i < n4) {
        float4 a = ((const float4*)p)[i];
        float4 b = ((const float4*)x0)[i];
        ushort4 o;
        o.x = f2bf(2.f * a.x - b.x); o.y = f2bf(2.f * a.y - b.y);
        o.z = f2bf(2.f * a.z - b.z); o.w = f2bf(2.f * a.w - b.w);
        ((ushort4*)y)[i] = o;
    }
}

// W [t][in][out] f32 -> Wt [t][out][in] bf16 (column-major per WMMA B-frag layout)
__global__ void wtrans_kernel(const float* __restrict__ W, unsigned short* __restrict__ Wt,
                              int kin, int nout, int total) {
    int idx = blockIdx.x * blockDim.x + threadIdx.x;
    if (idx >= total) return;
    int per = kin * nout;
    int t = idx / per;
    int r = idx - t * per;
    int i = r / nout;
    int o = r - i * nout;
    Wt[(size_t)t * per + (size_t)o * kin + i] = f2bf(W[idx]);
}

// ---------------- global mean pool ----------------
__global__ void cnt_kernel(const int* __restrict__ batch, float* __restrict__ cnt) {
    int n = blockIdx.x * blockDim.x + threadIdx.x;
    if (n < NODES) atomicAdd(&cnt[batch[n]], 1.0f);
}

// concat[g2(256) | feature(128)] -> segment-sum into sums[128][384]
__global__ void pool_kernel(const float* __restrict__ g2, const float* __restrict__ feat,
                            const int* __restrict__ batch, float* __restrict__ sums) {
    int idx = blockIdx.x * blockDim.x + threadIdx.x;
    if (idx >= NODES * 96) return;
    int n  = idx / 96;
    int c  = (idx - n * 96) << 2;           // 0..380 step 4
    float4 v = (c < 256) ? *(const float4*)(g2 + (size_t)n * 256 + c)
                         : *(const float4*)(feat + (size_t)n * 128 + (c - 256));
    float* o = sums + (size_t)batch[n] * 384 + c;
    atomicAdd(o + 0, v.x);
    atomicAdd(o + 1, v.y);
    atomicAdd(o + 2, v.z);
    atomicAdd(o + 3, v.w);
}

__global__ void pool_fin_kernel(const float* __restrict__ sums, const float* __restrict__ cnt,
                                unsigned short* __restrict__ pbf) {
    int idx = blockIdx.x * blockDim.x + threadIdx.x;
    if (idx >= NGRAPH * 384) return;
    int b = idx / 384;
    pbf[idx] = f2bf(sums[idx] / fmaxf(cnt[b], 1.0f));
}

// ---------------- WMMA GEMM: out = relu( sum_t A_t @ W_t^T + bias ) ----------------
// A_t: [M][K] bf16 row-major; Wt: [nt][N][K] bf16 (weight pre-transposed).
// One wave computes a 16x64 output strip (4 accumulators): the A-frag is loaded
// once per k-step and reused across 4 B-frags -> ~13 flops per L2 byte.
// Requires: K % 32 == 0, M % 16 == 0, N % 64 == 0.
__global__ __launch_bounds__(256) void wmma_gemm_kernel(
        const unsigned short* __restrict__ A0, const unsigned short* __restrict__ A1,
        const unsigned short* __restrict__ A2, const unsigned short* __restrict__ Wt,
        const float* __restrict__ bias, float* __restrict__ outf,
        unsigned short* __restrict__ outb, int M, int N, int K, int nt, int relu) {
    int wave   = blockIdx.x * (blockDim.x >> 5) + (threadIdx.x >> 5);
    int lane   = threadIdx.x & 31;
    int tilesN = N >> 6;                      // 64-wide strips
    int ntiles = (M >> 4) * tilesN;
    if (wave >= ntiles) return;               // uniform per-wave exit: EXEC stays all-1s
    int tm  = wave / tilesN;
    int tn  = wave - tm * tilesN;             // strip index (64 columns)
    int l16 = lane & 15;
    int hi  = lane >> 4;

    size_t arow_off = (size_t)((tm << 4) + l16) * K;          // A-frag row = lane&15
    size_t bcol_off = (size_t)((tn << 6) + l16) * K;          // first B column of strip

    v8f acc0 = {0.f,0.f,0.f,0.f,0.f,0.f,0.f,0.f};
    v8f acc1 = acc0, acc2 = acc0, acc3 = acc0;

    for (int t = 0; t < nt; ++t) {
        const unsigned short* A    = (t == 0) ? A0 : (t == 1) ? A1 : A2;
        const unsigned short* arow = A + arow_off;
        const unsigned short* bcol = Wt + (size_t)t * N * K + bcol_off;
        for (int k0 = 0; k0 < K; k0 += 32) {
            __builtin_prefetch(arow + k0 + 64, 0, 0);         // global_prefetch_b8
            union Frag { bf16x16 v; uint4 q[2]; };
            Frag af, b0, b1, b2, b3;
            // A 16x32 bf16 layout: lane<16 -> K {k0..k0+15}; lane>=16 -> K {k0+16..k0+31}
            af.q[0] = *(const uint4*)(arow + k0 + hi * 16);
            af.q[1] = *(const uint4*)(arow + k0 + hi * 16 + 8);
            // B 32x16 bf16 layout: lane<16 -> K {k0..k0+15}; lane>=16 -> K {k0+16..k0+31}
            size_t bk = (size_t)k0 + hi * 16;
            b0.q[0] = *(const uint4*)(bcol + bk);
            b0.q[1] = *(const uint4*)(bcol + bk + 8);
            b1.q[0] = *(const uint4*)(bcol + (size_t)16 * K + bk);
            b1.q[1] = *(const uint4*)(bcol + (size_t)16 * K + bk + 8);
            b2.q[0] = *(const uint4*)(bcol + (size_t)32 * K + bk);
            b2.q[1] = *(const uint4*)(bcol + (size_t)32 * K + bk + 8);
            b3.q[0] = *(const uint4*)(bcol + (size_t)48 * K + bk);
            b3.q[1] = *(const uint4*)(bcol + (size_t)48 * K + bk + 8);
            acc0 = __builtin_amdgcn_wmma_f32_16x16x32_bf16(false, af.v, false, b0.v, (short)0, acc0, false, false);
            acc1 = __builtin_amdgcn_wmma_f32_16x16x32_bf16(false, af.v, false, b1.v, (short)0, acc1, false, false);
            acc2 = __builtin_amdgcn_wmma_f32_16x16x32_bf16(false, af.v, false, b2.v, (short)0, acc2, false, false);
            acc3 = __builtin_amdgcn_wmma_f32_16x16x32_bf16(false, af.v, false, b3.v, (short)0, acc3, false, false);
        }
    }

#pragma unroll
    for (int j = 0; j < 4; ++j) {
        v8f acc = (j == 0) ? acc0 : (j == 1) ? acc1 : (j == 2) ? acc2 : acc3;
        int ncol = (tn << 6) + (j << 4) + l16;
        float bv = bias ? bias[ncol] : 0.f;
#pragma unroll
        for (int r = 0; r < 8; ++r) {
            float v = acc[r] + bv;
            if (relu) v = fmaxf(v, 0.f);
            size_t o = (size_t)((tm << 4) + r + (hi << 3)) * N + ncol;
            if (outf) outf[o] = v;
            if (outb) outb[o] = f2bf(v);
        }
    }
}

// ---------------- host orchestration ----------------
static inline int g256(long long n) { return (int)((n + 255) / 256); }
static inline int ggemm(int M, int N) { int nt = (M >> 4) * (N >> 6); return (nt + 7) / 8; }

extern "C" void kernel_launch(void* const* d_in, const int* in_sizes, int n_in,
                              void* d_out, int out_size, void* d_ws, size_t ws_size,
                              hipStream_t stream) {
    const float* feature = (const float*)d_in[0];
    const int*   src     = (const int*)d_in[1];
    const int*   dst     = src + EDGES;
    const int*   batch   = (const int*)d_in[2];
    const float* W1      = (const float*)d_in[3];
    const float* b1      = (const float*)d_in[4];
    const float* W2      = (const float*)d_in[5];
    const float* b2      = (const float*)d_in[6];
    const float* fc1w    = (const float*)d_in[7];
    const float* fc1b    = (const float*)d_in[8];
    const float* fc2w    = (const float*)d_in[9];
    const float* fc2b    = (const float*)d_in[10];
    (void)in_sizes; (void)n_in; (void)out_size; (void)ws_size;

    char*  base = (char*)d_ws;
    size_t off  = 0;
    auto alloc = [&](size_t bytes) -> void* {
        void* p = base + off;
        off += (bytes + 255) & ~(size_t)255;
        return p;
    };

    float*          dinv  = (float*)alloc((size_t)NODES * 4);
    float*          ew    = (float*)alloc((size_t)EDGES * 4);
    float*          t1f   = (float*)alloc((size_t)NODES * FEAT * 4);
    float*          pf    = (float*)alloc((size_t)NODES * FEAT * 4);
    unsigned short* xbf   = (unsigned short*)alloc((size_t)NODES * FEAT * 2);
    unsigned short* t1bf  = (unsigned short*)alloc((size_t)NODES * FEAT * 2);
    unsigned short* t2bf  = (unsigned short*)alloc((size_t)NODES * FEAT * 2);
    float*          g1f   = (float*)alloc((size_t)NODES * FEAT * 4);
    unsigned short* g1bf  = (unsigned short*)alloc((size_t)NODES * FEAT * 2);
    float*          g2f   = (float*)alloc((size_t)NODES * 2 * FEAT * 4);
    unsigned short* w1t   = (unsigned short*)alloc((size_t)3 * FEAT * FEAT * 2);
    unsigned short* w2t   = (unsigned short*)alloc((size_t)3 * FEAT * 2 * FEAT * 2);
    unsigned short* fc1t  = (unsigned short*)alloc((size_t)HID * 3 * FEAT * 2);
    unsigned short* fc2t  = (unsigned short*)alloc((size_t)NOUT * HID * 2);
    float*          pooled= (float*)alloc((size_t)NGRAPH * 3 * FEAT * 4);
    float*          cnt   = (float*)alloc((size_t)NGRAPH * 4);
    unsigned short* pbf   = (unsigned short*)alloc((size_t)NGRAPH * 3 * FEAT * 2);
    unsigned short* hbf   = (unsigned short*)alloc((size_t)NGRAPH * HID * 2);

    const int nfeat4 = NODES * FEAT / 4;

    // ---- edge weights ----
    hipMemsetAsync(dinv, 0, (size_t)NODES * 4, stream);
    deg_kernel  <<<g256(EDGES), 256, 0, stream>>>(src, dinv);
    dinv_kernel <<<g256(NODES), 256, 0, stream>>>(dinv);
    edgew_kernel<<<g256(EDGES), 256, 0, stream>>>(src, dst, dinv, ew);

    // ---- precision conversions (weights once, features once) ----
    cvt_bf16_kernel<<<g256(nfeat4), 256, 0, stream>>>(feature, xbf, nfeat4);
    wtrans_kernel<<<g256(3 * FEAT * FEAT),     256, 0, stream>>>(W1,   w1t,  FEAT,     FEAT,     3 * FEAT * FEAT);
    wtrans_kernel<<<g256(3 * FEAT * 2 * FEAT), 256, 0, stream>>>(W2,   w2t,  FEAT,     2 * FEAT, 3 * FEAT * 2 * FEAT);
    wtrans_kernel<<<g256(3 * FEAT * HID),      256, 0, stream>>>(fc1w, fc1t, 3 * FEAT, HID,      3 * FEAT * HID);
    wtrans_kernel<<<g256(HID * NOUT),          256, 0, stream>>>(fc2w, fc2t, HID,      NOUT,     HID * NOUT);

    // ---- ChebConv 1 ----
    hipMemsetAsync(t1f, 0, (size_t)NODES * FEAT * 4, stream);
    prop_kernel<<<g256((long long)EDGES * 32), 256, 0, stream>>>(feature, ew, src, dst, t1f);
    cvt_bf16_kernel<<<g256(nfeat4), 256, 0, stream>>>(t1f, t1bf, nfeat4);
    hipMemsetAsync(pf, 0, (size_t)NODES * FEAT * 4, stream);
    prop_kernel<<<g256((long long)EDGES * 32), 256, 0, stream>>>(t1f, ew, src, dst, pf);
    cheb2_kernel<<<g256(nfeat4), 256, 0, stream>>>(pf, feature, t2bf, nfeat4);
    wmma_gemm_kernel<<<ggemm(NODES, FEAT), 256, 0, stream>>>(
        xbf, t1bf, t2bf, w1t, b1, g1f, g1bf, NODES, FEAT, FEAT, 3, 1);

    // ---- ChebConv 2 ----
    hipMemsetAsync(t1f, 0, (size_t)NODES * FEAT * 4, stream);
    prop_kernel<<<g256((long long)EDGES * 32), 256, 0, stream>>>(g1f, ew, src, dst, t1f);
    cvt_bf16_kernel<<<g256(nfeat4), 256, 0, stream>>>(t1f, t1bf, nfeat4);
    hipMemsetAsync(pf, 0, (size_t)NODES * FEAT * 4, stream);
    prop_kernel<<<g256((long long)EDGES * 32), 256, 0, stream>>>(t1f, ew, src, dst, pf);
    cheb2_kernel<<<g256(nfeat4), 256, 0, stream>>>(pf, g1f, t2bf, nfeat4);
    wmma_gemm_kernel<<<ggemm(NODES, 2 * FEAT), 256, 0, stream>>>(
        g1bf, t1bf, t2bf, w2t, b2, g2f, (unsigned short*)nullptr, NODES, 2 * FEAT, FEAT, 3, 1);

    // ---- global mean pool over concat[g2 | feature] ----
    hipMemsetAsync(pooled, 0, (size_t)NGRAPH * 3 * FEAT * 4, stream);
    hipMemsetAsync(cnt,    0, (size_t)NGRAPH * 4, stream);
    cnt_kernel <<<g256(NODES), 256, 0, stream>>>(batch, cnt);
    pool_kernel<<<g256((long long)NODES * 96), 256, 0, stream>>>(g2f, feature, batch, pooled);
    pool_fin_kernel<<<g256(NGRAPH * 3 * FEAT), 256, 0, stream>>>(pooled, cnt, pbf);

    // ---- FC head ----
    wmma_gemm_kernel<<<ggemm(NGRAPH, HID), 256, 0, stream>>>(
        pbf, (unsigned short*)nullptr, (unsigned short*)nullptr, fc1t, fc1b,
        (float*)nullptr, hbf, NGRAPH, HID, 3 * FEAT, 1, 1);
    wmma_gemm_kernel<<<ggemm(NGRAPH, NOUT), 256, 0, stream>>>(
        hbf, (unsigned short*)nullptr, (unsigned short*)nullptr, fc2t, fc2b,
        (float*)d_out, (unsigned short*)nullptr, NGRAPH, NOUT, HID, 1, 0);
}
